// AdaptiveIncrementalAttention_51049981281333
// MI455X (gfx1250) — matrix operand order repaired and checked by reading, not compiled, not run on hardware
//
#include <hip/hip_runtime.h>

// Problem constants (match reference)
#define BB   8
#define NN   2048
#define DD   1024
#define LL   128

typedef __attribute__((ext_vector_type(16))) __bf16 v16bf;
typedef __attribute__((ext_vector_type(8)))  __bf16 v8bf;   // 16 bytes
typedef __attribute__((ext_vector_type(8)))  float  v8f;

#define LDS_STRIDE 1040   // 1024 halves + 16-half pad (de-aligns LDS banks)

// Branch-free tanh: one v_exp_f32 + a few VALU ops, exact saturation for |x| big.
__device__ __forceinline__ float fast_tanh(float x) {
  float ax = __builtin_fabsf(x);
  float u  = __expf(-2.0f * ax);          // in (0,1], no overflow
  float t  = __fdividef(1.0f - u, 1.0f + u);
  return __builtin_copysignf(t, x);
}

// ---------------------------------------------------------------------------
// K0: convert W_k (f32 row-major [D,D]) -> bf16 row-major
// ---------------------------------------------------------------------------
__global__ void aia_cvt_wk(const float* __restrict__ wk, __bf16* __restrict__ out) {
  int i = blockIdx.x * blockDim.x + threadIdx.x;
  if (i < DD * DD) out[i] = (__bf16)wk[i];
}

// ---------------------------------------------------------------------------
// K1: q[b,d] = sum_e X[b,e] * W[d,e]   (B*D threads, 1024-long dot, float4)
// ---------------------------------------------------------------------------
__global__ void aia_bd_dot(const float* __restrict__ X, const float* __restrict__ W,
                           float* __restrict__ out) {
  int t = blockIdx.x * blockDim.x + threadIdx.x;   // t < B*D
  int b = t >> 10, d = t & (DD - 1);
  const float4* xp = (const float4*)(X + (size_t)b * DD);
  const float4* wp = (const float4*)(W + (size_t)d * DD);
  float acc = 0.f;
  for (int e = 0; e < DD / 4; ++e) {
    float4 x = xp[e], w = wp[e];
    acc += x.x * w.x + x.y * w.y + x.z * w.z + x.w * w.w;
  }
  out[t] = acc;
}

// ---------------------------------------------------------------------------
// K2 (main): sigma[b,n] = omega . tanh(q[b] + e_j[b,n] @ W_k^T) ; write exp()
// 4 waves / WG, each wave owns a 16-row strip of flattened (b,n).
// bf16 WMMA 16x16x32, f32 accumulate. A staged in LDS (bf16), B from global.
// 4-way N-tile blocking: each A fragment read from LDS feeds 4 WMMAs.
// ---------------------------------------------------------------------------
__global__ void aia_sigma(const float* __restrict__ ej, const __bf16* __restrict__ wk,
                          const float* __restrict__ q,  const float* __restrict__ omega,
                          float* __restrict__ exps) {
  extern __shared__ __align__(16) unsigned short smem_raw[];
  const int lane = threadIdx.x & 31;
  const int wave = threadIdx.x >> 5;
  const int row0 = (blockIdx.x * 4 + wave) * 16;     // global row in [0, B*N)
  const int b    = row0 >> 11;                       // row0 / 2048 (strip stays in one b)
  __bf16* amem = (__bf16*)smem_raw + (size_t)wave * 16 * LDS_STRIDE;

  // ---- stage + convert this wave's 16x1024 f32 strip into bf16 LDS ----
  for (int r = 0; r < 16; ++r) {
    const float* src = ej + (size_t)(row0 + r) * DD;
    __bf16* dst = amem + r * LDS_STRIDE;
    for (int c = lane * 8; c < DD; c += 32 * 8) {
      float4 f0 = *(const float4*)(src + c);
      float4 f1 = *(const float4*)(src + c + 4);
      v8bf h;
      h[0] = (__bf16)f0.x; h[1] = (__bf16)f0.y; h[2] = (__bf16)f0.z; h[3] = (__bf16)f0.w;
      h[4] = (__bf16)f1.x; h[5] = (__bf16)f1.y; h[6] = (__bf16)f1.z; h[7] = (__bf16)f1.w;
      *(v8bf*)(dst + c) = h;
    }
  }
  __syncthreads();

  const int hi = lane >> 4;      // 0 or 1 (which K half for A / B fragments)
  const int lo = lane & 15;      // A: row within tile ; B/C: column within tile
  float accs[8] = {0.f, 0.f, 0.f, 0.f, 0.f, 0.f, 0.f, 0.f};
  const __bf16* arow = amem + lo * LDS_STRIDE + hi * 16;

  for (int nb = 0; nb < 16; ++nb) {                  // 16 iters x 4 N-tiles = 64
    const int n0 = nb * 64;
    v8f c0 = {}, c1 = {}, c2 = {}, c3 = {};
    const __bf16* br0 = wk + (size_t)(n0 +  0 + lo) * DD + hi * 16;
    const __bf16* br1 = wk + (size_t)(n0 + 16 + lo) * DD + hi * 16;
    const __bf16* br2 = wk + (size_t)(n0 + 32 + lo) * DD + hi * 16;
    const __bf16* br3 = wk + (size_t)(n0 + 48 + lo) * DD + hi * 16;
#pragma unroll 2
    for (int kk = 0; kk < DD; kk += 32) {
      // A fragment from LDS: K = kk + hi*16 + {0..15} contiguous
      const __bf16* ap = arow + kk;
      v16bf a;
      *(v8bf*)&a         = *(const v8bf*)ap;
      *(((v8bf*)&a) + 1) = *(const v8bf*)(ap + 8);
      // 4 B fragments from global bf16 W_k: K = kk + hi*16 + {0..15} contiguous
      v16bf b0, b1, b2, b3;
      *(v8bf*)&b0         = *(const v8bf*)(br0 + kk);
      *(((v8bf*)&b0) + 1) = *(const v8bf*)(br0 + kk + 8);
      *(v8bf*)&b1         = *(const v8bf*)(br1 + kk);
      *(((v8bf*)&b1) + 1) = *(const v8bf*)(br1 + kk + 8);
      *(v8bf*)&b2         = *(const v8bf*)(br2 + kk);
      *(((v8bf*)&b2) + 1) = *(const v8bf*)(br2 + kk + 8);
      *(v8bf*)&b3         = *(const v8bf*)(br3 + kk);
      *(((v8bf*)&b3) + 1) = *(const v8bf*)(br3 + kk + 8);
      c0 = __builtin_amdgcn_wmma_f32_16x16x32_bf16(false, a, false, b0, (short)0, c0, false, false);
      c1 = __builtin_amdgcn_wmma_f32_16x16x32_bf16(false, a, false, b1, (short)0, c1, false, false);
      c2 = __builtin_amdgcn_wmma_f32_16x16x32_bf16(false, a, false, b2, (short)0, c2, false, false);
      c3 = __builtin_amdgcn_wmma_f32_16x16x32_bf16(false, a, false, b3, (short)0, c3, false, false);
    }
    // C element (vgpr v, lane) = k_j[row0 + v + 8*hi][n_tile + lo]
#pragma unroll
    for (int t = 0; t < 4; ++t) {
      const v8f c = (t == 0) ? c0 : (t == 1) ? c1 : (t == 2) ? c2 : c3;
      const int n = n0 + t * 16 + lo;
      const float om = omega[n];
      const float qv = q[b * DD + n];
#pragma unroll
      for (int v = 0; v < 8; ++v)
        accs[v] += om * fast_tanh(c[v] + qv);
    }
  }

  // reduce each row's partial across its 16 N-lanes (xor stays within half-wave)
#pragma unroll
  for (int v = 0; v < 8; ++v) {
    float s = accs[v];
    for (int m = 1; m < 16; m <<= 1) s += __shfl_xor(s, m, 32);
    accs[v] = s;
  }
  if (lo == 0) {                       // lane 0 -> rows 0..7 ; lane 16 -> rows 8..15
    const int base = row0 + hi * 8;
#pragma unroll
    for (int v = 0; v < 8; ++v)
      exps[base + v] = __expf(accs[v]);
  }
}

// ---------------------------------------------------------------------------
// K3: deterministic per-b sum of exp_sigma over N (tree reduce, no atomics)
// ---------------------------------------------------------------------------
__global__ void aia_sum(const float* __restrict__ exps, float* __restrict__ sums) {
  __shared__ float red[256];
  const int b = blockIdx.x, t = threadIdx.x;
  float acc = 0.f;
  for (int n = t; n < NN; n += 256) acc += exps[b * NN + n];
  red[t] = acc;
  __syncthreads();
  for (int s = 128; s > 0; s >>= 1) {
    if (t < s) red[t] += red[t + s];
    __syncthreads();
  }
  if (t == 0) sums[b] = red[0];
}

// ---------------------------------------------------------------------------
// K4: a_ij = imp * exp / (sum + 1e-9)
// ---------------------------------------------------------------------------
__global__ void aia_aij(const float* __restrict__ imp, const float* __restrict__ exps,
                        const float* __restrict__ sums, float* __restrict__ aij) {
  int i = blockIdx.x * blockDim.x + threadIdx.x;     // i < B*N
  aij[i] = imp[i] * exps[i] / (sums[i >> 11] + 1e-9f);
}

// ---------------------------------------------------------------------------
// K5: s_part[b,slice,e] = sum_{n in slice} a[b,n] * e_j[b,n,e]
// 512 WGs (B x 4 e-chunks x 16 n-slices) -> enough WGPs in flight for HBM BW.
// ---------------------------------------------------------------------------
__global__ void aia_wsum(const float* __restrict__ aij, const float* __restrict__ ej,
                         float* __restrict__ s_part) {
  const int b     = blockIdx.x >> 6;
  const int chunk = (blockIdx.x >> 4) & 3;
  const int slice = blockIdx.x & 15;
  const int e     = (chunk << 8) + threadIdx.x;
  const int nbeg  = slice * (NN / 16);
  float acc = 0.f;
  const float* ejb = ej + ((size_t)b * NN + nbeg) * DD + e;
  const float* ab  = aij + b * NN + nbeg;
  for (int n = 0; n < NN / 16; ++n) acc += ab[n] * ejb[(size_t)n * DD];
  s_part[(size_t)(b * 16 + slice) * DD + e] = acc;
}

// K5b: s[b,e] = sum_slice s_part[b,slice,e]   (fixed-order, deterministic)
__global__ void aia_wsum_red(const float* __restrict__ s_part, float* __restrict__ s) {
  const int t = blockIdx.x * blockDim.x + threadIdx.x;   // t < B*D
  const int b = t >> 10, e = t & (DD - 1);
  float acc = 0.f;
#pragma unroll
  for (int sl = 0; sl < 16; ++sl) acc += s_part[(size_t)(b * 16 + sl) * DD + e];
  s[t] = acc;
}

// ---------------------------------------------------------------------------
// K6: A[b,d] = s[b,:] . W_v[d,:] ; write A and A_l
// ---------------------------------------------------------------------------
__global__ void aia_final(const float* __restrict__ s, const float* __restrict__ wv,
                          float* __restrict__ out) {
  const int t = blockIdx.x * blockDim.x + threadIdx.x;   // t < B*D
  const int b = t >> 10, d = t & (DD - 1);
  const float4* sp = (const float4*)(s + (size_t)b * DD);
  const float4* wp = (const float4*)(wv + (size_t)d * DD);
  float acc = 0.f;
  for (int e = 0; e < DD / 4; ++e) {
    float4 x = sp[e], w = wp[e];
    acc += x.x * w.x + x.y * w.y + x.z * w.z + x.w * w.w;
  }
  const size_t AL_OFF = (size_t)BB * DD + (size_t)BB * LL * DD;
  out[t]          = acc;     // A
  out[AL_OFF + t] = acc;     // A_l (identical)
}

// K7: A_lk[b,l,d] = A[b,d] * R[l,d]   (4096 WGs, fully coalesced broadcast)
__global__ void aia_alk(const float* __restrict__ R, float* __restrict__ out) {
  const int i = blockIdx.x * blockDim.x + threadIdx.x;   // i < B*L*D
  const int b = i >> 17;                                 // / (L*D)
  const int l = (i >> 10) & (LL - 1);
  const int d = i & (DD - 1);
  out[(size_t)BB * DD + i] = out[(b << 10) + d] * R[l * DD + d];
}

// ---------------------------------------------------------------------------
extern "C" void kernel_launch(void* const* d_in, const int* in_sizes, int n_in,
                              void* d_out, int out_size, void* d_ws, size_t ws_size,
                              hipStream_t stream) {
  const float* e_prev = (const float*)d_in[0];   // [B,D]
  const float* e_j    = (const float*)d_in[1];   // [B,N,D]
  const float* imp    = (const float*)d_in[2];   // [B,N]
  const float* R      = (const float*)d_in[3];   // [L,D]
  const float* W_q    = (const float*)d_in[4];   // [D,D]
  const float* W_k    = (const float*)d_in[5];   // [D,D]
  const float* W_v    = (const float*)d_in[6];   // [D,D]
  const float* omega  = (const float*)d_in[7];   // [D]
  float* out = (float*)d_out;

  char* ws = (char*)d_ws;
  __bf16* wk_bf  = (__bf16*)(ws);                       // 2 MB
  float*  q      = (float*)(ws + 2097152);              // 32 KB
  float*  exps   = (float*)(ws + 2129920);              // 64 KB
  float*  aij    = (float*)(ws + 2195456);              // 64 KB
  float*  sums   = (float*)(ws + 2260992);              // 64 B
  float*  s      = (float*)(ws + 2261056);              // 32 KB
  float*  s_part = (float*)(ws + 2293824);              // 512 KB

  // K0: W_k -> bf16
  aia_cvt_wk<<<(DD * DD) / 256, 256, 0, stream>>>(W_k, wk_bf);
  // K1: q = e_prev @ W_q^T
  aia_bd_dot<<<(BB * DD) / 256, 256, 0, stream>>>(e_prev, W_q, q);
  // K2: main bf16 WMMA sigma kernel (dynamic LDS: 4 waves * 16 * 1040 * 2 bytes)
  aia_sigma<<<(BB * NN) / 64, 128, 4 * 16 * LDS_STRIDE * 2, stream>>>(
      e_j, wk_bf, q, omega, exps);
  // K3: per-b sums (deterministic)
  aia_sum<<<BB, 256, 0, stream>>>(exps, sums);
  // K4: attention weights
  aia_aij<<<(BB * NN) / 256, 256, 0, stream>>>(imp, exps, sums, aij);
  // K5: weighted token sum, 16 N-slices for parallelism, then fixed-order reduce
  aia_wsum<<<BB * 4 * 16, 256, 0, stream>>>(aij, e_j, s_part);
  aia_wsum_red<<<(BB * DD) / 256, 256, 0, stream>>>(s_part, s);
  // K6: A = s @ W_v^T -> A and A_l
  aia_final<<<(BB * DD) / 256, 256, 0, stream>>>(s, W_v, out);
  // K7: A_lk = A (x) R broadcast
  aia_alk<<<(BB * LL * DD) / 256, 256, 0, stream>>>(R, out);
}
